// GAT_21457656611528
// MI455X (gfx1250) — matrix operand order repaired and checked by reading, not compiled, not run on hardware
//
#include <hip/hip_runtime.h>
#include <hip/hip_bf16.h>

// ---------------------------------------------------------------------------
// Problem constants (from reference setup_inputs)
#define NNODE  4096
#define F0     512
#define NHEAD  4
#define ALPHA  0.2f
#define NEG_INF_F (-9.0e15f)

typedef __attribute__((ext_vector_type(16))) __bf16       v16bf;
typedef __attribute__((ext_vector_type(8)))  float        v8f;
typedef __attribute__((ext_vector_type(8)))  unsigned int v8u;

static __device__ __forceinline__ unsigned short f2bf(float f) {
    unsigned int u = __builtin_bit_cast(unsigned int, f);
    unsigned int r = u + 0x7FFFu + ((u >> 16) & 1u);   // round-to-nearest-even
    return (unsigned short)(r >> 16);
}
static __device__ __forceinline__ unsigned int pack2bf(float a, float b) {
    return (unsigned int)f2bf(a) | ((unsigned int)f2bf(b) << 16);
}
static __device__ __forceinline__ float lrelu(float e) {
    return e > 0.f ? e : ALPHA * e;
}

// ---------------------------------------------------------------------------
// f32 -> bf16 cast, row-major (for the A operand: x)
__global__ void cast_bf16_kernel(const float* __restrict__ in,
                                 unsigned short* __restrict__ out, int n) {
    int i = blockIdx.x * blockDim.x + threadIdx.x;
    if (i < n) out[i] = f2bf(in[i]);
}

// f32 [H,K,O] -> bf16 [H,O,K] (transposed weights: B-operand fragments become
// two contiguous 16B loads per lane)
__global__ void cast_transpose_w_kernel(const float* __restrict__ in,
                                        unsigned short* __restrict__ out,
                                        int K, int O, int H) {
    int i = blockIdx.x * blockDim.x + threadIdx.x;
    int tot = H * K * O;
    if (i >= tot) return;
    int h = i / (K * O);
    int r = i % (K * O);
    int k = r / O, o = r % O;
    out[((size_t)h * O + o) * K + k] = f2bf(in[i]);
}

// ---------------------------------------------------------------------------
// Wh[h] = Xb @ W[h]  (bf16 WMMA, f32 accumulate). One wave per 16x16 tile.
// Xb:[N,K] bf16 row-major. Wt:[H,O,K] bf16 (transposed). Outputs:
//   Whf:[H,N,O] f32 (for f1/f2)  and  WhT:[H,O,N] bf16 (B operand of agg).
__global__ void gemm_wh_kernel(const unsigned short* __restrict__ Xb,
                               const unsigned short* __restrict__ Wt,
                               float* __restrict__ Whf,
                               unsigned short* __restrict__ WhT,
                               int Nn, int K, int O, int H) {
    int wave = (int)((blockIdx.x * blockDim.x + threadIdx.x) >> 5);
    int lane = threadIdx.x & 31;
    int colTiles = O >> 4;
    int perHead  = (Nn >> 4) * colTiles;
    if (wave >= H * perHead) return;
    int h  = wave / perHead;
    int t  = wave % perHead;
    int rt = t / colTiles, ct = t % colTiles;
    int half = lane >> 4, l = lane & 15;
    int row = rt * 16 + l;          // A row for this lane
    int col = ct * 16 + l;          // B/D column for this lane

    const unsigned short* ar = Xb + (size_t)row * K;
    const unsigned short* br = Wt + ((size_t)h * O + col) * K;

    v8f acc = {};
    for (int kk = 0; kk < K; kk += 32) {
        // A: lane needs K runs [kk+half*8, +8) and [kk+16+half*8, +8)
        const uint4 a0 = *(const uint4*)(ar + kk + half * 8);
        const uint4 a1 = *(const uint4*)(ar + kk + 16 + half * 8);
        v8u ap = {a0.x, a0.y, a0.z, a0.w, a1.x, a1.y, a1.z, a1.w};
        // B: lane needs K runs [kk+half*8, +8) and [kk+16+half*8, +8) at fixed column (transposed layout)
        const uint4 b0 = *(const uint4*)(br + kk + half * 8);
        const uint4 b1 = *(const uint4*)(br + kk + 16 + half * 8);
        v8u bp = {b0.x, b0.y, b0.z, b0.w, b1.x, b1.y, b1.z, b1.w};
        acc = __builtin_amdgcn_wmma_f32_16x16x32_bf16(
                false, __builtin_bit_cast(v16bf, ap),
                false, __builtin_bit_cast(v16bf, bp),
                (short)0, acc, false, false);
    }
    // f32 row-major (scattered, small) for f1/f2
#pragma unroll
    for (int r = 0; r < 8; ++r) {
        int orow = rt * 16 + half * 8 + r;     // D: M = r + 8*half
        Whf[((size_t)h * Nn + orow) * O + col] = acc[r];
    }
    // bf16 transposed: 8 consecutive rows at fixed column -> one 16B store
    uint4 wp;
    wp.x = pack2bf(acc[0], acc[1]);
    wp.y = pack2bf(acc[2], acc[3]);
    wp.z = pack2bf(acc[4], acc[5]);
    wp.w = pack2bf(acc[6], acc[7]);
    *(uint4*)(WhT + ((size_t)h * O + col) * Nn + rt * 16 + half * 8) = wp;
}

// ---------------------------------------------------------------------------
// f1[h,n] = Wh[h,n,:] . a[h,:O] ; f2[h,n] = Wh[h,n,:] . a[h,O:]
__global__ void f1f2_kernel(const float* __restrict__ Whf,
                            const float* __restrict__ a,
                            float* __restrict__ f1, float* __restrict__ f2,
                            int Nn, int O, int H) {
    int idx = blockIdx.x * blockDim.x + threadIdx.x;
    if (idx >= H * Nn) return;
    int h = idx / Nn, n = idx % Nn;
    const float* wr = Whf + ((size_t)h * Nn + n) * O;
    const float* ah = a + (size_t)h * 2 * O;
    float s1 = 0.f, s2 = 0.f;
    for (int o = 0; o < O; o += 4) {
        float4 w = *(const float4*)(wr + o);
        float4 u = *(const float4*)(ah + o);
        float4 v = *(const float4*)(ah + O + o);
        s1 += w.x * u.x + w.y * u.y + w.z * u.z + w.w * u.w;
        s2 += w.x * v.x + w.y * v.y + w.z * v.z + w.w * v.w;
    }
    f1[idx] = s1; f2[idx] = s2;
}

// ---------------------------------------------------------------------------
// Pass 1 of fused softmax: per row i, ALL heads at once (adj read once).
// One wave per row; lanes take 4 consecutive j (int4/float4); shuffle reduce.
__global__ void stats_kernel(const int* __restrict__ adj,
                             const float* __restrict__ f1,
                             const float* __restrict__ f2,
                             float* __restrict__ m, float* __restrict__ dinv,
                             int Nn) {
    int wave = (int)((blockIdx.x * blockDim.x + threadIdx.x) >> 5);
    int lane = threadIdx.x & 31;
    if (wave >= Nn) return;
    const int i = wave;
    const int* arow = adj + (size_t)i * Nn;
    float fi[NHEAD], mx[NHEAD], ss[NHEAD];
#pragma unroll
    for (int h = 0; h < NHEAD; ++h) { fi[h] = f1[h * Nn + i]; mx[h] = -3.0e38f; }

    for (int jb = lane * 4; jb < Nn; jb += 128) {
        int4 A = *(const int4*)(arow + jb);
#pragma unroll
        for (int h = 0; h < NHEAD; ++h) {
            float4 F = *(const float4*)(f2 + h * Nn + jb);
            float e0 = A.x ? lrelu(fi[h] + F.x) : NEG_INF_F;
            float e1 = A.y ? lrelu(fi[h] + F.y) : NEG_INF_F;
            float e2 = A.z ? lrelu(fi[h] + F.z) : NEG_INF_F;
            float e3 = A.w ? lrelu(fi[h] + F.w) : NEG_INF_F;
            mx[h] = fmaxf(mx[h], fmaxf(fmaxf(e0, e1), fmaxf(e2, e3)));
        }
    }
#pragma unroll
    for (int h = 0; h < NHEAD; ++h) {
#pragma unroll
        for (int off = 16; off; off >>= 1)
            mx[h] = fmaxf(mx[h], __shfl_xor(mx[h], off, 32));
        ss[h] = 0.f;
    }
    for (int jb = lane * 4; jb < Nn; jb += 128) {
        int4 A = *(const int4*)(arow + jb);
#pragma unroll
        for (int h = 0; h < NHEAD; ++h) {
            float4 F = *(const float4*)(f2 + h * Nn + jb);
            float e0 = A.x ? lrelu(fi[h] + F.x) : NEG_INF_F;
            float e1 = A.y ? lrelu(fi[h] + F.y) : NEG_INF_F;
            float e2 = A.z ? lrelu(fi[h] + F.z) : NEG_INF_F;
            float e3 = A.w ? lrelu(fi[h] + F.w) : NEG_INF_F;
            ss[h] += __expf(e0 - mx[h]) + __expf(e1 - mx[h]) +
                     __expf(e2 - mx[h]) + __expf(e3 - mx[h]);
        }
    }
#pragma unroll
    for (int h = 0; h < NHEAD; ++h) {
#pragma unroll
        for (int off = 16; off; off >>= 1) ss[h] += __shfl_xor(ss[h], off, 32);
        if (lane == 0) { m[h * Nn + i] = mx[h]; dinv[h * Nn + i] = 1.0f / ss[h]; }
    }
}

// ---------------------------------------------------------------------------
// Pass 2: h[i,:] = sum_j softmax(e_ij) * Wh[j,:], fused ELU + head concat.
// Template OT = O/16. One wave builds the 16x32 P tile ONCE per K-chunk and
// feeds OT accumulators. Split-K by 2 (stats precomputed -> partials add
// linearly); partner partials combined through LDS. Attention matrix is never
// materialized. Grids are sized exactly (full blocks) so __syncthreads is safe.
template <int OT>
__global__ void agg_kernel(const int* __restrict__ adj,
                           const float* __restrict__ f1,
                           const float* __restrict__ f2,
                           const float* __restrict__ m,
                           const float* __restrict__ dinv,
                           const unsigned short* __restrict__ WhT,
                           float* __restrict__ outF,
                           unsigned short* __restrict__ outB,
                           int Nn, int H) {
    const int O = OT * 16;
    __shared__ float red[4][OT * 256];     // 4 wave-pairs per 256-thread block

    int gw   = (int)((blockIdx.x * blockDim.x + threadIdx.x) >> 5);
    int lane = threadIdx.x & 31;
    int wIb  = threadIdx.x >> 5;           // wave in block (0..7)
    int pair = wIb >> 1;                   // wave-pair in block (0..3)
    int js   = gw & 1;                     // K-split half
    int tile = gw >> 1;                    // (h, rowTile)
    int rowTiles = Nn >> 4;
    int h  = tile / rowTiles;
    int rt = tile % rowTiles;
    int half = lane >> 4, l = lane & 15;
    int row = rt * 16 + l;

    float fi = f1[h * Nn + row];
    float mi = m [h * Nn + row];
    float di = dinv[h * Nn + row];
    const float* f2h  = f2 + h * Nn;
    const int*   arow = adj + (size_t)row * Nn;
    const unsigned short* Bb = WhT + (size_t)h * O * Nn;

    v8f acc[OT];
#pragma unroll
    for (int ot = 0; ot < OT; ++ot) acc[ot] = (v8f){};

    const int jBeg = js * (Nn >> 1), jEnd = jBeg + (Nn >> 1);
    for (int j0 = jBeg; j0 < jEnd; j0 += 32) {
        __builtin_prefetch(arow + j0 + 256, 0, 1);   // global_prefetch_b8
        // Build P tile (A operand) once: 4 quads of 4 consecutive j
        v8u ap;
#pragma unroll
        for (int q = 0; q < 4; ++q) {
            int jq = j0 + ((q & 2) ? 16 : 0) + half * 8 + (q & 1) * 4;
            int4   A = *(const int4*)  (arow + jq);
            float4 F = *(const float4*)(f2h  + jq);
            float e0 = A.x ? lrelu(fi + F.x) : NEG_INF_F;
            float e1 = A.y ? lrelu(fi + F.y) : NEG_INF_F;
            float e2 = A.z ? lrelu(fi + F.z) : NEG_INF_F;
            float e3 = A.w ? lrelu(fi + F.w) : NEG_INF_F;
            ap[q * 2]     = pack2bf(__expf(e0 - mi) * di, __expf(e1 - mi) * di);
            ap[q * 2 + 1] = pack2bf(__expf(e2 - mi) * di, __expf(e3 - mi) * di);
        }
        v16bf a = __builtin_bit_cast(v16bf, ap);
#pragma unroll
        for (int ot = 0; ot < OT; ++ot) {
            const unsigned short* bt = Bb + (size_t)(ot * 16 + l) * Nn + j0 + half * 8;
            const uint4 b0 = *(const uint4*)bt;
            const uint4 b1 = *(const uint4*)(bt + 16);
            v8u bp = {b0.x, b0.y, b0.z, b0.w, b1.x, b1.y, b1.z, b1.w};
            acc[ot] = __builtin_amdgcn_wmma_f32_16x16x32_bf16(
                          false, a, false, __builtin_bit_cast(v16bf, bp),
                          (short)0, acc[ot], false, false);
        }
    }

    // combine split-K partials through LDS
    if (js == 1) {
#pragma unroll
        for (int ot = 0; ot < OT; ++ot)
#pragma unroll
            for (int r = 0; r < 8; ++r)
                red[pair][ot * 256 + r * 32 + lane] = acc[ot][r];
    }
    __syncthreads();
    if (js == 0) {
        int HO = H * O;
#pragma unroll
        for (int ot = 0; ot < OT; ++ot) {
            int col = ot * 16 + l;
#pragma unroll
            for (int r = 0; r < 8; ++r) {
                float v = acc[ot][r] + red[pair][ot * 256 + r * 32 + lane];
                float ev = v > 0.f ? v : (__expf(v) - 1.f);      // ELU
                int orow = rt * 16 + half * 8 + r;
                size_t idx = (size_t)orow * HO + (size_t)h * O + col;
                outF[idx] = ev;
                outB[idx] = f2bf(ev);
            }
        }
    }
}

// ---------------------------------------------------------------------------
// logits = H @ Wlin + b ; out = log_softmax(logits). One wave per node row.
__global__ void final_kernel(const float* __restrict__ Hf,
                             const float* __restrict__ Wlin,
                             const float* __restrict__ blin,
                             float* __restrict__ out, int Nn, int K, int C) {
    int wave = (int)((blockIdx.x * blockDim.x + threadIdx.x) >> 5);
    int lane = threadIdx.x & 31;
    if (wave >= Nn) return;
    const float* hr = Hf + (size_t)wave * K;
    int c0 = lane, c1 = lane + 32;
    float l0 = (c0 < C) ? blin[c0] : -3.0e38f;
    float l1 = (c1 < C) ? blin[c1] : -3.0e38f;
    for (int k = 0; k < K; ++k) {
        float hv = hr[k];
        if (c0 < C) l0 += hv * Wlin[k * C + c0];
        if (c1 < C) l1 += hv * Wlin[k * C + c1];
    }
    float mx = fmaxf(l0, l1);
#pragma unroll
    for (int off = 16; off; off >>= 1) mx = fmaxf(mx, __shfl_xor(mx, off, 32));
    float s = ((c0 < C) ? __expf(l0 - mx) : 0.f) + ((c1 < C) ? __expf(l1 - mx) : 0.f);
#pragma unroll
    for (int off = 16; off; off >>= 1) s += __shfl_xor(s, off, 32);
    float lse = mx + __logf(s);
    if (c0 < C) out[(size_t)wave * C + c0] = l0 - lse;
    if (c1 < C) out[(size_t)wave * C + c1] = l1 - lse;
}

// ---------------------------------------------------------------------------
extern "C" void kernel_launch(void* const* d_in, const int* in_sizes, int n_in,
                              void* d_out, int out_size, void* d_ws, size_t ws_size,
                              hipStream_t stream) {
    const float* x    = (const float*)d_in[0];
    const int*   adj  = (const int*)  d_in[1];
    const float* W1   = (const float*)d_in[2];
    const float* a1   = (const float*)d_in[3];
    const float* W2   = (const float*)d_in[4];
    const float* a2   = (const float*)d_in[5];
    const float* W3   = (const float*)d_in[6];
    const float* a3   = (const float*)d_in[7];
    const float* Wlin = (const float*)d_in[8];
    const float* blin = (const float*)d_in[9];
    float* out = (float*)d_out;

    const int N = NNODE, H = NHEAD;

    // ---- workspace carve-up (256B aligned; ~18.5 MB total) -----------------
    char* ws = (char*)d_ws;
    size_t off = 0;
    auto carve = [&](size_t bytes) -> char* {
        char* p = ws + off;
        off = (off + bytes + 255) & ~(size_t)255;
        return p;
    };
    unsigned short* xb  = (unsigned short*)carve((size_t)N * F0 * 2);        // 4 MB
    unsigned short* h1b = (unsigned short*)carve((size_t)N * 256 * 2);       // 2 MB
    unsigned short* h2b = (unsigned short*)carve((size_t)N * 128 * 2);       // 1 MB
    unsigned short* h3b = (unsigned short*)carve((size_t)N * 64 * 2);        // .5 MB
    unsigned short* W1t = (unsigned short*)carve((size_t)H * F0 * 64 * 2);   // 256 KB
    unsigned short* W2t = (unsigned short*)carve((size_t)H * 256 * 32 * 2);  // 64 KB
    unsigned short* W3t = (unsigned short*)carve((size_t)H * 128 * 16 * 2);  // 16 KB
    float*          Whf = (float*)         carve((size_t)H * N * 64 * 4);    // 4 MB
    unsigned short* WhT = (unsigned short*)carve((size_t)H * 64 * N * 2);    // 2 MB
    float*          f1  = (float*)carve((size_t)H * N * 4);
    float*          f2  = (float*)carve((size_t)H * N * 4);
    float*          mrow= (float*)carve((size_t)H * N * 4);
    float*          drow= (float*)carve((size_t)H * N * 4);
    float*          hF  = (float*)carve((size_t)N * 256 * 4);                // 4 MB
    (void)ws_size; (void)in_sizes; (void)n_in; (void)out_size;

    cast_bf16_kernel<<<(N * F0 + 255) / 256, 256, 0, stream>>>(x, xb, N * F0);
    cast_transpose_w_kernel<<<(H * F0 * 64 + 255) / 256, 256, 0, stream>>>(W1, W1t, F0,  64, H);
    cast_transpose_w_kernel<<<(H * 256 * 32 + 255) / 256, 256, 0, stream>>>(W2, W2t, 256, 32, H);
    cast_transpose_w_kernel<<<(H * 128 * 16 + 255) / 256, 256, 0, stream>>>(W3, W3t, 128, 16, H);

    const int statsBlocks = N * 32 / 256;          // one wave per row, exact
    const int aggBlocks   = H * (N / 16) * 2 * 32 / 256;   // split-K=2, exact

    auto pre = [&](const unsigned short* inb, const unsigned short* Wt,
                   const float* avec, int K, int O) {
        int tiles = H * (N / 16) * (O / 16);
        gemm_wh_kernel<<<tiles * 32 / 256, 256, 0, stream>>>(inb, Wt, Whf, WhT, N, K, O, H);
        f1f2_kernel<<<(H * N + 255) / 256, 256, 0, stream>>>(Whf, avec, f1, f2, N, O, H);
        stats_kernel<<<statsBlocks, 256, 0, stream>>>(adj, f1, f2, mrow, drow, N);
    };

    // stage 1: [N,512] -> [N,256]
    pre(xb, W1t, a1, F0, 64);
    agg_kernel<4><<<aggBlocks, 256, 0, stream>>>(adj, f1, f2, mrow, drow, WhT, hF, h1b, N, H);
    // stage 2: [N,256] -> [N,128]
    pre(h1b, W2t, a2, 256, 32);
    agg_kernel<2><<<aggBlocks, 256, 0, stream>>>(adj, f1, f2, mrow, drow, WhT, hF, h2b, N, H);
    // stage 3: [N,128] -> [N,64]  (hF keeps the f32 copy for the classifier)
    pre(h2b, W3t, a3, 128, 16);
    agg_kernel<1><<<aggBlocks, 256, 0, stream>>>(adj, f1, f2, mrow, drow, WhT, hF, h3b, N, H);

    final_kernel<<<N * 32 / 256, 256, 0, stream>>>(hF, Wlin, blin, out, N, 64, 40);
}
